// MPNNLayer_90134183674510
// MI455X (gfx1250) — compile-verified, matrix operand order checked by reading, not run on hardware
//
#include <hip/hip_runtime.h>
#include <math.h>

// ---------------------------------------------------------------------------
// MPNN layer for MI455X (gfx1250, wave32).
// H=64, N_NODES=12000, N_EDGES=50000.
// Dominant cost: eh[E,128] @ W2^T[128,4096] (52.4 GFLOP), fused with the
// per-edge contraction against node[src] so Wedge (819 MB) never hits HBM.
// GEMMs on v_wmma_f32_16x16x32_bf16 (f32 accumulate). W2 slabs are staged
// into LDS with global_load_async_to_lds_b128 (double-buffered, shared by
// the 4 waves of a workgroup), overlapping fill with WMMA compute.
// ---------------------------------------------------------------------------

#define H 64
#define NN 12000
#define NE 50000
#define NT 3125   // NE/16: number of 16-edge tiles

typedef __bf16 v8bf  __attribute__((ext_vector_type(8)));
typedef __bf16 v16bf __attribute__((ext_vector_type(16)));
typedef float  v8f   __attribute__((ext_vector_type(8)));

// float -> bf16, round to nearest even
__device__ __forceinline__ __bf16 f2bf(float f) {
  union { float f; unsigned u; } v; v.f = f;
  unsigned r = v.u + 0x7FFFu + ((v.u >> 16) & 1u);
  unsigned short h = (unsigned short)(r >> 16);
  __bf16 b; __builtin_memcpy(&b, &h, 2);
  return b;
}

// Load one 16x32 bf16 WMMA operand tile (A or B) from a row-major matrix
// (works for global or LDS-backed pointers).  `half` = lane>>4 selects the
// K-subset per the CDNA5 16-bit operand layout (K 0..7/16..23 vs 8..15/24..31).
__device__ __forceinline__ v16bf ld_tile(const __bf16* base, int row, int stride,
                                         int k0, int half) {
  const __bf16* p = base + (size_t)row * stride + k0 + (half << 3);
  v8bf c0 = *(const v8bf*)(p);
  v8bf c1 = *(const v8bf*)(p + 16);
  return __builtin_shufflevector(c0, c1, 0,1,2,3,4,5,6,7,8,9,10,11,12,13,14,15);
}

__device__ __forceinline__ v8f wmma_bf16(v16bf a, v16bf b, v8f c) {
  return __builtin_amdgcn_wmma_f32_16x16x32_bf16(false, a, false, b, (short)0, c,
                                                 false, false);
}

// ---------------------------------------------------------------------------
__global__ __launch_bounds__(256) void k_cvt_bf16(const float* __restrict__ in,
                                                  __bf16* __restrict__ out, int n) {
  int i = blockIdx.x * blockDim.x + threadIdx.x;
  if (i < n) out[i] = f2bf(in[i]);
}

__global__ __launch_bounds__(256) void k_relu_cvt(const float* __restrict__ in,
                                                  __bf16* __restrict__ out, int n) {
  int i = blockIdx.x * blockDim.x + threadIdx.x;
  if (i < n) { float v = in[i]; out[i] = f2bf(v > 0.f ? v : 0.f); }
}

__global__ __launch_bounds__(256) void k_zero(float* __restrict__ p, int n) {
  int i = blockIdx.x * blockDim.x + threadIdx.x;
  if (i < n) p[i] = 0.f;
}

// ---------------------------------------------------------------------------
// eh = relu(edge @ W1^T + b1), bf16 in, bf16 out.  M=NE, N=128, K=64.
__global__ __launch_bounds__(256) void k_edge_mlp1(const __bf16* __restrict__ edge_b,
                                                   const __bf16* __restrict__ W1b,
                                                   const float* __restrict__ b1,
                                                   __bf16* __restrict__ eh, int E) {
  const int gw   = (blockIdx.x * blockDim.x + threadIdx.x) >> 5;
  const int lane = threadIdx.x & 31;
  const int mt = gw >> 3, nt = gw & 7;
  const int m0 = mt << 4;
  if (m0 >= E) return;
  const int half = lane >> 4, r = lane & 15;
  const int n0 = nt << 4;
  v8f acc = {0.f,0.f,0.f,0.f,0.f,0.f,0.f,0.f};
  acc = wmma_bf16(ld_tile(edge_b, m0 + r, 64, 0,  half),
                  ld_tile(W1b,    n0 + r, 64, 0,  half), acc);
  acc = wmma_bf16(ld_tile(edge_b, m0 + r, 64, 32, half),
                  ld_tile(W1b,    n0 + r, 64, 32, half), acc);
  const float bias = b1[n0 + r];
#pragma unroll
  for (int j = 0; j < 8; ++j) {
    float v = acc[j] + bias;
    v = v > 0.f ? v : 0.f;
    eh[(size_t)(m0 + j + (half << 3)) * 128 + n0 + r] = f2bf(v);
  }
}

// ---------------------------------------------------------------------------
// Fused msg kernel. One wave per 16-edge tile, 4 waves per workgroup.
// Per hb in [0,64): the 4 n-tiles n = hb*64 + kt*16 consume a contiguous
// 16 KB slab of W2 (rows hb*64..hb*64+63, 128 cols).  The workgroup
// double-buffers that slab in LDS via global_load_async_to_lds_b128 and
// overlaps the next fill with the current 16 WMMAs per wave.
__global__ __launch_bounds__(128) void k_msg(const __bf16* __restrict__ ehb,
                                             const __bf16* __restrict__ W2b,
                                             const float* __restrict__ b2,
                                             const float* __restrict__ node,
                                             const int* __restrict__ src,
                                             const int* __restrict__ dst,
                                             float* __restrict__ agg) {
  __shared__ __align__(16) __bf16 wslab[2][64 * 128];  // 2 x 16 KB W2 slabs
  __shared__ float xsh[4][16 * 64];                    // gathered node[src]
  __shared__ int   dsh[4][16];
  const int wv = threadIdx.x >> 5, lane = threadIdx.x & 31;
  const int tid = threadIdx.x;
  const int t  = blockIdx.x * 4 + wv;                  // 16-edge tile id
  const int tc = t < NT ? t : NT - 1;                  // clamped (tail WG)
  const bool valid = t < NT;
  const int e0 = tc << 4;
  float* xs = xsh[wv];
  int*   ds = dsh[wv];

  if (lane < 16) ds[lane] = dst[e0 + lane];
  for (int i = lane; i < 16 * 64; i += 32) {           // gather src rows (f32)
    const int rr = i >> 6, cc = i & 63;
    xs[i] = node[(size_t)src[e0 + rr] * 64 + cc];
  }

  const int half = lane >> 4, r = lane & 15;

  v16bf A[4];                                          // eh block [16x128] bf16
#pragma unroll
  for (int ck = 0; ck < 4; ++ck) A[ck] = ld_tile(ehb, e0 + r, 128, ck << 5, half);

  // prologue: async fill of slab 0 (128 threads x 16B x 8 rounds = 16 KB)
  {
    const unsigned la = (unsigned)(uintptr_t)(&wslab[0][0]) + tid * 16;
    const unsigned long long ga = (unsigned long long)(W2b + (size_t)tid * 8);
#pragma unroll
    for (int q = 0; q < 8; ++q)
      asm volatile("global_load_async_to_lds_b128 %0, %1, off offset:%2"
                   :: "v"(la), "v"(ga), "i"(q * 2048) : "memory");
  }

  const v8f z8 = {0.f,0.f,0.f,0.f,0.f,0.f,0.f,0.f};
  v8f msgacc[4] = {z8, z8, z8, z8};                    // msg[16 edges, 64 k]
  float xv[8];

  for (int hb = 0; hb < 64; ++hb) {
    const int cur = hb & 1;
    asm volatile("s_wait_asynccnt 0x0" ::: "memory");  // own fills complete
    __syncthreads();                                   // slab[cur] resident
    if (hb < 63) {                                     // kick next slab fill
      const unsigned la =
          (unsigned)(uintptr_t)(&wslab[cur ^ 1][0]) + tid * 16;
      const unsigned long long ga = (unsigned long long)(
          W2b + (size_t)(hb + 1) * (64 * 128) + (size_t)tid * 8);
#pragma unroll
      for (int q = 0; q < 8; ++q)
        asm volatile("global_load_async_to_lds_b128 %0, %1, off offset:%2"
                     :: "v"(la), "v"(ga), "i"(q * 2048) : "memory");
    }
#pragma unroll
    for (int j = 0; j < 8; ++j)
      xv[j] = xs[((j + (half << 3)) << 6) + hb];       // x[e = j+8*half, hb]
    const __bf16* wb = &wslab[cur][0];
#pragma unroll
    for (int kt = 0; kt < 4; ++kt) {
      v8f d = z8;
#pragma unroll
      for (int ck = 0; ck < 4; ++ck)                   // K=128: 4 bf16 WMMAs
        d = wmma_bf16(A[ck], ld_tile(wb, (kt << 4) + r, 128, ck << 5, half), d);
      const float b2v = b2[(hb << 6) + (kt << 4) + r];
      v8f acc = msgacc[kt];
#pragma unroll
      for (int j = 0; j < 8; ++j) acc[j] += xv[j] * (d[j] + b2v);
      msgacc[kt] = acc;
    }
    __syncthreads();                                   // done reading slab[cur]
  }

  if (valid) {                                         // scatter-add to agg[dst]
#pragma unroll
    for (int kt = 0; kt < 4; ++kt) {
      v8f acc = msgacc[kt];
#pragma unroll
      for (int j = 0; j < 8; ++j) {
        const int e = j + (half << 3);
        float* p = agg + (size_t)ds[e] * 64 + (kt << 4) + r;
        (void)__hip_atomic_fetch_add(p, acc[j], __ATOMIC_RELAXED,
                                     __HIP_MEMORY_SCOPE_AGENT);
      }
    }
  }
}

// ---------------------------------------------------------------------------
// out[M,192] = X[M,64] @ W^T (W row-major [192,64]).  One wave per 16x16 tile.
__global__ __launch_bounds__(256) void k_gemm_gru(const __bf16* __restrict__ X,
                                                  const __bf16* __restrict__ W,
                                                  float* __restrict__ out, int M) {
  const int gw   = (blockIdx.x * blockDim.x + threadIdx.x) >> 5;
  const int lane = threadIdx.x & 31;
  const int mt = gw / 12, nt = gw % 12;
  const int m0 = mt << 4;
  if (m0 >= M) return;
  const int half = lane >> 4, r = lane & 15;
  const int n0 = nt << 4;
  v8f acc = {0.f,0.f,0.f,0.f,0.f,0.f,0.f,0.f};
  acc = wmma_bf16(ld_tile(X, m0 + r, 64, 0,  half),
                  ld_tile(W, n0 + r, 64, 0,  half), acc);
  acc = wmma_bf16(ld_tile(X, m0 + r, 64, 32, half),
                  ld_tile(W, n0 + r, 64, 32, half), acc);
#pragma unroll
  for (int j = 0; j < 8; ++j)
    out[(size_t)(m0 + j + (half << 3)) * 192 + n0 + r] = acc[j];
}

// ---------------------------------------------------------------------------
// GRU gates + LayerNorm. One wave per node; each lane owns 2 of 64 columns.
__device__ __forceinline__ float sigm(float x) { return 1.f / (1.f + expf(-x)); }

__global__ __launch_bounds__(256) void k_gru_ln(const float* __restrict__ gi,
                                                const float* __restrict__ gh,
                                                const float* __restrict__ bih,
                                                const float* __restrict__ bhh,
                                                const float* __restrict__ hidden,
                                                const float* __restrict__ gamma,
                                                const float* __restrict__ beta,
                                                float* __restrict__ out, int M) {
  const int wv = threadIdx.x >> 5, lane = threadIdx.x & 31;
  const int nd = blockIdx.x * 8 + wv;
  if (nd >= M) return;
  const float* gir = gi + (size_t)nd * 192;
  const float* ghr = gh + (size_t)nd * 192;
  float o[2];
#pragma unroll
  for (int t = 0; t < 2; ++t) {
    const int c = lane + t * 32;
    const float rg = sigm(gir[c]       + bih[c]       + ghr[c]       + bhh[c]);
    const float zg = sigm(gir[64 + c]  + bih[64 + c]  + ghr[64 + c]  + bhh[64 + c]);
    const float ng = tanhf(gir[128 + c] + bih[128 + c] +
                           rg * (ghr[128 + c] + bhh[128 + c]));
    const float hv = hidden[(size_t)nd * 64 + c];
    o[t] = (1.f - zg) * ng + zg * hv;
  }
  float s = o[0] + o[1], s2 = o[0] * o[0] + o[1] * o[1];
#pragma unroll
  for (int off = 16; off > 0; off >>= 1) {
    s  += __shfl_xor(s,  off);
    s2 += __shfl_xor(s2, off);
  }
  const float mu  = s * (1.f / 64.f);
  const float var = s2 * (1.f / 64.f) - mu * mu;
  const float inv = rsqrtf(var + 1e-5f);
#pragma unroll
  for (int t = 0; t < 2; ++t) {
    const int c = lane + t * 32;
    out[(size_t)nd * 64 + c] = (o[t] - mu) * inv * gamma[c] + beta[c];
  }
}

// ---------------------------------------------------------------------------
extern "C" void kernel_launch(void* const* d_in, const int* in_sizes, int n_in,
                              void* d_out, int out_size, void* d_ws, size_t ws_size,
                              hipStream_t stream) {
  const float* node   = (const float*)d_in[0];
  const float* edge   = (const float*)d_in[1];
  const float* hidden = (const float*)d_in[2];
  const int*   src    = (const int*)d_in[3];
  const int*   dst    = (const int*)d_in[4];
  const float* W1     = (const float*)d_in[5];
  const float* b1     = (const float*)d_in[6];
  const float* W2     = (const float*)d_in[7];
  const float* b2     = (const float*)d_in[8];
  const float* W_ih   = (const float*)d_in[9];
  const float* W_hh   = (const float*)d_in[10];
  const float* b_ih   = (const float*)d_in[11];
  const float* b_hh   = (const float*)d_in[12];
  const float* gamma  = (const float*)d_in[13];
  const float* beta   = (const float*)d_in[14];
  float* out = (float*)d_out;

  char* ws = (char*)d_ws;
  size_t off = 0;
  auto alloc = [&](size_t bytes) -> char* {
    off = (off + 255) & ~(size_t)255;
    char* p = ws + off;
    off += bytes;
    return p;
  };
  __bf16* edge_b = (__bf16*)alloc((size_t)NE * 64 * 2);
  __bf16* W1b    = (__bf16*)alloc((size_t)128 * 64 * 2);
  __bf16* W2b    = (__bf16*)alloc((size_t)4096 * 128 * 2);
  __bf16* Wihb   = (__bf16*)alloc((size_t)192 * 64 * 2);
  __bf16* Whhb   = (__bf16*)alloc((size_t)192 * 64 * 2);
  __bf16* hidb   = (__bf16*)alloc((size_t)NN * 64 * 2);
  __bf16* ehb    = (__bf16*)alloc((size_t)NE * 128 * 2);
  float*  agg    = (float*)alloc((size_t)NN * 64 * 4);
  __bf16* xnb    = (__bf16*)alloc((size_t)NN * 64 * 2);
  float*  gi     = (float*)alloc((size_t)NN * 192 * 4);
  float*  gh     = (float*)alloc((size_t)NN * 192 * 4);
  (void)ws_size; (void)in_sizes; (void)n_in; (void)out_size;

  auto cvt = [&](const float* in, __bf16* o, int n) {
    k_cvt_bf16<<<(n + 255) / 256, 256, 0, stream>>>(in, o, n);
  };
  cvt(edge, edge_b, NE * 64);
  cvt(W1, W1b, 128 * 64);
  cvt(W2, W2b, 4096 * 128);
  cvt(W_ih, Wihb, 192 * 64);
  cvt(W_hh, Whhb, 192 * 64);
  cvt(hidden, hidb, NN * 64);

  k_zero<<<(NN * 64 + 255) / 256, 256, 0, stream>>>(agg, NN * 64);

  // eh = relu(edge @ W1^T + b1):  (NE/16)*8 tiles, 8 waves / 256-thread block
  k_edge_mlp1<<<NE / 16, 256, 0, stream>>>(edge_b, W1b, b1, ehb, NE);

  // fused big GEMM + gather/contract/scatter (4 tiles per WG)
  k_msg<<<(NT + 3) / 4, 128, 0, stream>>>(ehb, W2b, b2, node, src, dst, agg);

  // x_new = bf16(relu(agg))
  k_relu_cvt<<<(NN * 64 + 255) / 256, 256, 0, stream>>>(agg, xnb, NN * 64);

  // gi = x_new @ W_ih^T ; gh = hidden @ W_hh^T   (750*12 tiles each)
  k_gemm_gru<<<(NN / 16) * 12 / 8, 256, 0, stream>>>(xnb, Wihb, gi, NN);
  k_gemm_gru<<<(NN / 16) * 12 / 8, 256, 0, stream>>>(hidb, Whhb, gh, NN);

  // gates + layernorm
  k_gru_ln<<<(NN + 7) / 8, 256, 0, stream>>>(gi, gh, b_ih, b_hh, hidden,
                                             gamma, beta, out, NN);
}
